// PromptDurationMemoryEncoder_26697516712415
// MI455X (gfx1250) — compile-verified, matrix operand order checked by reading, not run on hardware
//
#include <hip/hip_runtime.h>
#include <math.h>

// ---- problem constants (match reference) ----
#define BB 64
#define TTT 4096
#define DD 64
#define MMM 64
#define NEGV -10000.0f

typedef __attribute__((ext_vector_type(16))) _Float16 v16h;
typedef __attribute__((ext_vector_type(8)))  _Float16 v8h;
typedef __attribute__((ext_vector_type(8)))  float    v8f;

__device__ __forceinline__ float gelu_exact(float x) {
  return 0.5f * x * (1.0f + erff(x * 0.70710678118654752f));
}

#define WMMA_F16(A, B, C) \
  __builtin_amdgcn_wmma_f32_16x16x32_f16(false, (A), false, (B), (short)0, (C), false, false)

#define CAT16(lo, hi) \
  __builtin_shufflevector((lo), (hi), 0, 1, 2, 3, 4, 5, 6, 7, 8, 9, 10, 11, 12, 13, 14, 15)

// ------------------------------------------------------------------
// Kernel A: mask, logdur, masked median (bitonic sort in LDS),
//           ref_residual, support. One block (256 thr) per batch row.
// ------------------------------------------------------------------
__global__ void k_median(const float* __restrict__ dur, const float* __restrict__ pmask,
                         float* __restrict__ o_mask, float* __restrict__ o_logdur,
                         float* __restrict__ o_resid, float* __restrict__ o_grate,
                         float* __restrict__ w_support) {
  __shared__ float s[TTT];
  __shared__ float red[256];
  __shared__ float fred[256];
  int b = blockIdx.x, tid = threadIdx.x;
  int cnt_local = 0; float msum_local = 0.f;
  for (int t = tid; t < TTT; t += 256) {
    float m  = fminf(fmaxf(pmask[(long long)b * TTT + t], 0.f), 1.f);
    float ld = logf(fmaxf(dur[(long long)b * TTT + t], 1e-4f)) * m;
    o_mask[(long long)b * TTT + t]   = m;
    o_logdur[(long long)b * TTT + t] = ld;
    bool valid = m > 0.5f;
    s[t] = valid ? ld : 1e30f;
    cnt_local += valid ? 1 : 0;
    msum_local += m;
  }
  red[tid] = (float)cnt_local; fred[tid] = msum_local;
  __syncthreads();
  for (int off = 128; off > 0; off >>= 1) {
    if (tid < off) { red[tid] += red[tid + off]; fred[tid] += fred[tid + off]; }
    __syncthreads();
  }
  int cnt = (int)(red[0] + 0.5f);
  float supraw = fred[0];
  if (tid == 0) w_support[b] = supraw;
  __syncthreads();
  for (int k = 2; k <= TTT; k <<= 1) {
    for (int j = k >> 1; j > 0; j >>= 1) {
      for (int i = tid; i < TTT; i += 256) {
        int ixj = i ^ j;
        if (ixj > i) {
          bool up = ((i & k) == 0);
          float a = s[i], c = s[ixj];
          if ((a > c) == up) { s[i] = c; s[ixj] = a; }
        }
      }
      __syncthreads();
    }
  }
  int idx = (cnt - 1) / 2;
  if (idx < 0) idx = 0;
  if (idx > TTT - 1) idx = TTT - 1;
  float med = (cnt > 0) ? s[idx] : 0.f;
  if (tid == 0) o_grate[b] = med;
  for (int t = tid; t < TTT; t += 256) {
    float m  = o_mask[(long long)b * TTT + t];
    float ld = o_logdur[(long long)b * TTT + t];
    o_resid[(long long)b * TTT + t] = (ld - med) * m;
  }
}

// ------------------------------------------------------------------
// Prep: f16 transposed conv weights wT[n][kg] = w[kg][n] (kg = tap*64+din)
//       and f16 role_key (already [n][k] row-major = what B needs).
// ------------------------------------------------------------------
__global__ void k_prep(const float* __restrict__ c1w, const float* __restrict__ c2w,
                       const float* __restrict__ role_key,
                       _Float16* __restrict__ w1T, _Float16* __restrict__ w2T,
                       _Float16* __restrict__ rkT) {
  int i = blockIdx.x * 256 + threadIdx.x;
  if (i < 12288) {
    int n = i / 192, kg = i % 192;
    w1T[i] = (_Float16)c1w[kg * 64 + n];
  } else if (i < 24576) {
    int j = i - 12288;
    int n = j / 192, kg = j % 192;
    w2T[j] = (_Float16)c2w[kg * 64 + n];
  } else if (i < 28672) {
    int j = i - 24576;
    rkT[j] = (_Float16)role_key[j];
  }
}

// ------------------------------------------------------------------
// Kernel B: hidden0 = emb[id] + resid*Wa[:,0] + ba  -> f16
// ------------------------------------------------------------------
__global__ void k_embed(const int* __restrict__ ids, const float* __restrict__ emb,
                        const float* __restrict__ Wa, const float* __restrict__ ba,
                        const float* __restrict__ resid, _Float16* __restrict__ bufA) {
  long long idx = (long long)blockIdx.x * 256 + threadIdx.x;  // B*T*D
  int d = (int)(idx % DD);
  long long bt = idx / DD;
  int id = ids[bt];
  float r = resid[bt];
  float h = emb[(long long)id * DD + d] + r * Wa[d * 4 + 0] + ba[d];
  bufA[idx] = (_Float16)h;
}

// ------------------------------------------------------------------
// Kernel C: causal K=3 conv as LDS-staged WMMA GEMM + exact GELU.
// Block = 256 thr (8 waves), tile = 64(T) x 64(D); each wave does two
// 16x16 tiles, K=192 -> 6 chained WMMAs each. Fragments are b128 LDS
// loads (two contiguous 8-half K-runs per lane).
// ------------------------------------------------------------------
__global__ void k_conv(const _Float16* __restrict__ xin, const _Float16* __restrict__ wT,
                       const float* __restrict__ bias,
                       _Float16* __restrict__ out16, float* __restrict__ out32, int mode) {
  __shared__ __align__(16) _Float16 xs[66 * 64];    // rows t0-2 .. t0+63
  __shared__ __align__(16) _Float16 wsm[64 * 192];  // wT staged
  __shared__ float yt[64 * 64];
  int b  = blockIdx.x >> 6;
  int t0 = (blockIdx.x & 63) * 64;
  int tid = threadIdx.x;
  // stage x tile (coalesced dword loads; zero-fill causal pad rows)
  const unsigned* xg = (const unsigned*)(xin + (long long)b * TTT * DD);  // 32 dwords/row
  unsigned* xs32 = (unsigned*)xs;
  for (int i = tid; i < 66 * 32; i += 256) {
    int srow = i >> 5, c = i & 31;
    int t = t0 - 2 + srow;
    xs32[i] = (t >= 0) ? xg[t * 32 + c] : 0u;
  }
  // stage weights (coalesced)
  const unsigned* wg = (const unsigned*)wT;
  unsigned* ws32 = (unsigned*)wsm;
  for (int i = tid; i < 6144; i += 256) ws32[i] = wg[i];
  __syncthreads();

  int wave = tid >> 5, lane = tid & 31;
  int g = lane >> 4, mr = lane & 15, nc = lane & 15;
  #pragma unroll
  for (int it = 0; it < 2; it++) {
    int ti = wave * 2 + it;
    int m0 = (ti >> 2) * 16;
    int nb = (ti & 3) * 16;
    v8f acc;
    float bv = bias[nb + nc];
    #pragma unroll
    for (int r = 0; r < 8; r++) acc[r] = bv;
    #pragma unroll
    for (int kk = 0; kk < 6; kk++) {
      int kg0 = kk * 32 + 8 * g;   // first 8-run
      int kg1 = kg0 + 16;          // second 8-run
      int tap0 = kg0 >> 6, din0 = kg0 & 63;
      int tap1 = kg1 >> 6, din1 = kg1 & 63;
      v8h a0 = *(const v8h*)&xs[(m0 + mr + tap0) * 64 + din0];
      v8h a1 = *(const v8h*)&xs[(m0 + mr + tap1) * 64 + din1];
      v8h b0 = *(const v8h*)&wsm[(nb + nc) * 192 + kg0];
      v8h b1 = *(const v8h*)&wsm[(nb + nc) * 192 + kg1];
      acc = WMMA_F16(CAT16(a0, a1), CAT16(b0, b1), acc);
    }
    #pragma unroll
    for (int r = 0; r < 8; r++) yt[(m0 + r + 8 * g) * 64 + nb + nc] = gelu_exact(acc[r]);
  }
  __syncthreads();
  long long gbase = ((long long)b * TTT + t0) * DD;
  if (mode == 0) {
    for (int i = tid; i < 4096; i += 256) out16[gbase + i] = (_Float16)yt[i];
  } else {
    for (int i = tid; i < 4096; i += 256) out32[gbase + i] = yt[i];
  }
}

// ------------------------------------------------------------------
// Kernel E: LayerNorm over D=64 + mask; one wave per row, shfl reduce.
// ------------------------------------------------------------------
__global__ void k_ln(float* __restrict__ bufC, _Float16* __restrict__ bufA,
                     const float* __restrict__ lng, const float* __restrict__ lnb,
                     const float* __restrict__ o_mask) {
  long long row = (long long)blockIdx.x * 8 + (threadIdx.x >> 5);
  int l = threadIdx.x & 31;
  float x0 = bufC[row * DD + l];
  float x1 = bufC[row * DD + 32 + l];
  float s = x0 + x1;
  for (int off = 16; off > 0; off >>= 1) s += __shfl_xor(s, off, 32);
  float mu = s * (1.f / 64.f);
  float d0 = x0 - mu, d1 = x1 - mu;
  float q = d0 * d0 + d1 * d1;
  for (int off = 16; off > 0; off >>= 1) q += __shfl_xor(q, off, 32);
  float inv = rsqrtf(q * (1.f / 64.f) + 1e-5f);
  float mv = o_mask[row];
  float y0 = (d0 * inv * lng[l] + lnb[l]) * mv;
  float y1 = (d1 * inv * lng[l + 32] + lnb[l + 32]) * mv;
  bufC[row * DD + l]      = y0;
  bufC[row * DD + 32 + l] = y1;
  bufA[row * DD + l]      = (_Float16)y0;
  bufA[row * DD + 32 + l] = (_Float16)y1;
}

// ------------------------------------------------------------------
// Kernel F: masked mean/std over T per (b,d). One block per b.
// ------------------------------------------------------------------
__global__ void k_meanstd(const float* __restrict__ bufC, const float* __restrict__ w_support,
                          float* __restrict__ w_mean, float* __restrict__ w_std) {
  int b = blockIdx.x;
  int d = threadIdx.x & 63;
  int chunk = threadIdx.x >> 6;  // 0..3
  __shared__ float s1[256], s2[256];
  float a1 = 0.f, a2 = 0.f;
  for (int t = chunk * 1024; t < (chunk + 1) * 1024; t++) {
    float h = bufC[((long long)b * TTT + t) * DD + d];
    a1 += h; a2 += h * h;
  }
  s1[threadIdx.x] = a1; s2[threadIdx.x] = a2; __syncthreads();
  if (chunk == 0) {
    float S1 = s1[d] + s1[d + 64] + s1[d + 128] + s1[d + 192];
    float S2 = s2[d] + s2[d + 64] + s2[d + 128] + s2[d + 192];
    float supraw = w_support[b];
    float den = fmaxf(supraw, 1.f);
    float mean = S1 / den;
    float var = (S2 - 2.f * mean * S1 + mean * mean * supraw) / den;
    w_mean[b * DD + d] = mean;
    w_std[b * DD + d]  = sqrtf(var + 1e-6f);
  }
}

// ------------------------------------------------------------------
// Kernel G: tiny MLP -> summary_state, coeff_norm. One block per b.
// ------------------------------------------------------------------
__global__ void k_mlp(const float* __restrict__ w_mean, const float* __restrict__ w_std,
                      const float* __restrict__ W1, const float* __restrict__ b1,
                      const float* __restrict__ W2, const float* __restrict__ b2,
                      const float* __restrict__ w_support,
                      float* __restrict__ o_summary, float* __restrict__ o_cnorm) {
  int b = blockIdx.x; int j = threadIdx.x;  // 64 threads
  __shared__ float cat[128];
  __shared__ float h1[64];
  __shared__ float red[64];
  cat[j]      = w_mean[b * DD + j];
  cat[j + 64] = w_std[b * DD + j];
  __syncthreads();
  float acc = b1[j];
  for (int k2 = 0; k2 < 128; k2++) acc += cat[k2] * W1[j * 128 + k2];
  h1[j] = gelu_exact(acc);
  __syncthreads();
  float acc2 = b2[j];
  for (int k2 = 0; k2 < 64; k2++) acc2 += h1[k2] * W2[j * 64 + k2];
  float s = tanhf(acc2);
  if (!(w_support[b] > 0.f)) s = 0.f;
  o_summary[b * DD + j] = s;
  red[j] = s * s; __syncthreads();
  for (int off = 32; off > 0; off >>= 1) { if (j < off) red[j] += red[j + off]; __syncthreads(); }
  if (j == 0) o_cnorm[b] = sqrtf(red[0]);
}

__global__ void k_zero(float* __restrict__ p, int n) {
  int i = blockIdx.x * 256 + threadIdx.x;
  if (i < n) p[i] = 0.f;
}

// ------------------------------------------------------------------
// Kernel H: score GEMM (LDS-staged WMMA), softmax over M, attn out
// (coalesced), one-pass role accumulators. Block = 256 thr (8 waves),
// tile = 32(T) x 64(M); one 16x16 tile per wave, 2 chained WMMAs.
// ------------------------------------------------------------------
__global__ void k_score(const _Float16* __restrict__ hid16, const _Float16* __restrict__ rkT,
                        const float* __restrict__ o_mask, const float* __restrict__ o_resid,
                        float* __restrict__ o_attn,
                        float* __restrict__ w_mass, float* __restrict__ w_s1,
                        float* __restrict__ w_s2) {
  __shared__ __align__(16) _Float16 hs[32 * 64];
  __shared__ __align__(16) _Float16 rk[64 * 64];
  __shared__ float sc[32 * 64];
  __shared__ float rres[32];
  __shared__ float rmask[32];
  int b  = blockIdx.x >> 7;
  int t0 = (blockIdx.x & 127) * 32;
  int tid = threadIdx.x;
  const unsigned* hg = (const unsigned*)(hid16 + ((long long)b * TTT + t0) * DD);
  unsigned* hs32 = (unsigned*)hs;
  for (int i = tid; i < 1024; i += 256) hs32[i] = hg[i];
  const unsigned* rg = (const unsigned*)rkT;
  unsigned* rk32 = (unsigned*)rk;
  for (int i = tid; i < 2048; i += 256) rk32[i] = rg[i];
  if (tid < 32) {
    rres[tid]  = o_resid[(long long)b * TTT + t0 + tid];
    rmask[tid] = o_mask[(long long)b * TTT + t0 + tid];
  }
  __syncthreads();

  int wave = tid >> 5, lane = tid & 31;
  int g = lane >> 4, mr = lane & 15, nc = lane & 15;
  int m0 = (wave >> 2) * 16;
  int nb = (wave & 3) * 16;
  v8f acc;
  #pragma unroll
  for (int r = 0; r < 8; r++) acc[r] = 0.f;
  #pragma unroll
  for (int kk = 0; kk < 2; kk++) {
    int kg0 = kk * 32 + 8 * g;
    int kg1 = kg0 + 16;
    v8h a0 = *(const v8h*)&hs[(m0 + mr) * 64 + kg0];
    v8h a1 = *(const v8h*)&hs[(m0 + mr) * 64 + kg1];
    v8h b0 = *(const v8h*)&rk[(nb + nc) * 64 + kg0];
    v8h b1 = *(const v8h*)&rk[(nb + nc) * 64 + kg1];
    acc = WMMA_F16(CAT16(a0, a1), CAT16(b0, b1), acc);
  }
  #pragma unroll
  for (int r = 0; r < 8; r++) sc[(m0 + r + 8 * g) * 64 + nb + nc] = acc[r] * 0.125f;
  __syncthreads();

  if (tid < 32) {
    int row = tid;
    float mv = rmask[row];
    float mx = -1e30f;
    for (int m = 0; m < 64; m++) {
      float v = (mv <= 0.f) ? NEGV : sc[row * 64 + m];
      sc[row * 64 + m] = v;
      mx = fmaxf(mx, v);
    }
    float ssum = 0.f;
    for (int m = 0; m < 64; m++) { float e = expf(sc[row * 64 + m] - mx); sc[row * 64 + m] = e; ssum += e; }
    float inv = 1.f / ssum;
    for (int m = 0; m < 64; m++) sc[row * 64 + m] *= inv * mv;
  }
  __syncthreads();
  // coalesced attn writeback
  float* og = o_attn + ((long long)b * TTT + t0) * MMM;
  for (int i = tid; i < 2048; i += 256) og[i] = sc[i];
  // one-pass role accumulators
  if (tid < 64) {
    int m = tid;
    float am = 0.f, a1 = 0.f, a2 = 0.f;
    for (int row = 0; row < 32; row++) {
      float a = sc[row * 64 + m];
      float r = rres[row];
      am += a; a1 += a * r; a2 += a * r * r;
    }
    atomicAdd(&w_mass[b * MMM + m], am);
    atomicAdd(&w_s1[b * MMM + m], a1);
    atomicAdd(&w_s2[b * MMM + m], a2);
  }
}

// ------------------------------------------------------------------
// Kernel I: finalize role_value / role_var / role_coverage.
// ------------------------------------------------------------------
__global__ void k_roles(const float* __restrict__ w_mass, const float* __restrict__ w_s1,
                        const float* __restrict__ w_s2, const float* __restrict__ w_support,
                        float* __restrict__ o_rv, float* __restrict__ o_rvar,
                        float* __restrict__ o_rcov) {
  int i = blockIdx.x * 256 + threadIdx.x;  // B*M
  int b = i / MMM;
  float mass = w_mass[i], S1 = w_s1[i], S2 = w_s2[i];
  float den = fmaxf(mass, 1e-6f);
  float v = S1 / den;
  float var = S2 / den - 2.f * v * (S1 / den) + v * v * (mass / den);
  var = (mass > 0.f) ? var : 1e-4f;
  var = fmaxf(var, 1e-4f);
  float sup = fmaxf(w_support[b], 1.f);
  float cov = fmaxf(mass / sup, 0.05f);
  o_rv[i] = v; o_rvar[i] = var; o_rcov[i] = cov;
}

// ------------------------------------------------------------------
// Kernel J: prompt_role_fit = (attn @ role_value) * mask.
// Block = 256 thr = 4 rows x 64 lanes (coalesced attn reads).
// ------------------------------------------------------------------
__global__ void k_fit(const float* __restrict__ o_attn, const float* __restrict__ o_rv,
                      const float* __restrict__ o_mask, float* __restrict__ o_fit) {
  __shared__ float red[256];
  long long row = (long long)blockIdx.x * 4 + (threadIdx.x >> 6);
  int m = threadIdx.x & 63;
  int b = (int)(row / TTT);
  float v = o_attn[row * MMM + m] * o_rv[(long long)b * MMM + m];
  red[threadIdx.x] = v; __syncthreads();
  for (int off = 32; off > 0; off >>= 1) {
    if (m < off) red[threadIdx.x] += red[threadIdx.x + off];
    __syncthreads();
  }
  if (m == 0) o_fit[row] = red[threadIdx.x] * o_mask[row];
}

// ------------------------------------------------------------------
extern "C" void kernel_launch(void* const* d_in, const int* in_sizes, int n_in,
                              void* d_out, int out_size, void* d_ws, size_t ws_size,
                              hipStream_t stream) {
  (void)in_sizes; (void)n_in; (void)out_size; (void)ws_size;
  const int*   unit_ids = (const int*)d_in[0];
  const float* dur      = (const float*)d_in[1];
  const float* pmask    = (const float*)d_in[2];
  const float* emb      = (const float*)d_in[3];
  const float* Wa       = (const float*)d_in[4];
  const float* ba       = (const float*)d_in[5];
  const float* c1w      = (const float*)d_in[6];
  const float* c1b      = (const float*)d_in[7];
  const float* c2w      = (const float*)d_in[8];
  const float* c2b      = (const float*)d_in[9];
  const float* ln_g     = (const float*)d_in[10];
  const float* ln_b     = (const float*)d_in[11];
  const float* W1       = (const float*)d_in[12];
  const float* b1       = (const float*)d_in[13];
  const float* W2       = (const float*)d_in[14];
  const float* b2       = (const float*)d_in[15];
  const float* role_key = (const float*)d_in[16];

  float* out = (float*)d_out;
  float* o_grate   = out + 0;            // [B,1]
  float* o_summary = out + 64;           // [B,D]
  float* o_rvalue  = out + 4160;         // [B,M]
  float* o_rvar    = out + 8256;         // [B,M]
  float* o_rcov    = out + 12352;        // [B,M]
  float* o_mask    = out + 16448;        // [B,T]
  float* o_logdur  = out + 278592;       // [B,T]
  float* o_resid   = out + 540736;       // [B,T]
  float* o_attn    = out + 802880;       // [B,T,M]
  float* o_fit     = out + 17580096;     // [B,T]
  float* o_cnorm   = out + 17842240;     // [B,1]

  char* ws = (char*)d_ws;
  const long long BTD = (long long)BB * TTT * DD;
  _Float16* bufA = (_Float16*)(ws + 0);        // f16 [B,T,D] (hid0 -> hidden f16)
  _Float16* bufB = (_Float16*)(ws + BTD * 2);  // f16 [B,T,D] (conv1 out)
  float*    bufC = (float*)(ws + BTD * 4);     // f32 [B,T,D] (conv2 out -> hidden)
  char* small = ws + BTD * 8;
  float*    w_support = (float*)(small + 0);        // [B]
  float*    w_mean    = (float*)(small + 256);      // [B,D]
  float*    w_std     = (float*)(small + 16640);    // [B,D]
  float*    w_mass    = (float*)(small + 33024);    // [B,M]
  float*    w_s1      = (float*)(small + 49408);    // [B,M]
  float*    w_s2      = (float*)(small + 65792);    // [B,M]
  _Float16* w1T       = (_Float16*)(small + 82176); // [64][192] f16
  _Float16* w2T       = (_Float16*)(small + 106752);// [64][192] f16
  _Float16* rkT       = (_Float16*)(small + 131328);// [64][64]  f16

  k_median<<<BB, 256, 0, stream>>>(dur, pmask, o_mask, o_logdur, o_resid, o_grate, w_support);
  k_prep<<<112, 256, 0, stream>>>(c1w, c2w, role_key, w1T, w2T, rkT);
  k_embed<<<(unsigned)(BTD / 256), 256, 0, stream>>>(unit_ids, emb, Wa, ba, o_resid, bufA);
  int conv_blocks = BB * (TTT / 64);  // 4096
  k_conv<<<conv_blocks, 256, 0, stream>>>(bufA, w1T, c1b, bufB, (float*)nullptr, 0);
  k_conv<<<conv_blocks, 256, 0, stream>>>(bufB, w2T, c2b, (_Float16*)nullptr, bufC, 1);
  k_ln<<<(BB * TTT) / 8, 256, 0, stream>>>(bufC, bufA, ln_g, ln_b, o_mask);
  k_meanstd<<<BB, 256, 0, stream>>>(bufC, w_support, w_mean, w_std);
  k_mlp<<<BB, 64, 0, stream>>>(w_mean, w_std, W1, b1, W2, b2, w_support, o_summary, o_cnorm);
  k_zero<<<(3 * BB * MMM + 255) / 256, 256, 0, stream>>>(w_mass, 3 * BB * MMM);
  k_score<<<BB * (TTT / 32), 256, 0, stream>>>(bufA, rkT, o_mask, o_resid, o_attn,
                                               w_mass, w_s1, w_s2);
  k_roles<<<(BB * MMM) / 256, 256, 0, stream>>>(w_mass, w_s1, w_s2, w_support,
                                                o_rvalue, o_rvar, o_rcov);
  k_fit<<<(BB * TTT) / 4, 256, 0, stream>>>(o_attn, o_rvalue, o_mask, o_fit);
}
